// CryoInit_26688926777635
// MI455X (gfx1250) — matrix operand order repaired and checked by reading, not compiled, not run on hardware
//
#include <hip/hip_runtime.h>
#include <stdint.h>

#define CEILDIV(a,b) (((a)+(b)-1)/(b))

typedef _Float16 h16;
typedef __attribute__((ext_vector_type(16))) _Float16 v16h;
typedef __attribute__((ext_vector_type(8)))  float    v8f;

#define NN     40      // nodes
#define KNBR   32      // neighbors
#define GRD    96      // cryo grid dim
#define CUBE   23
#define RECTZ  17
#define CEMB   256
#define CPLN   64
#define CEDGE  128
#define LNDIM  2176    // 128*17
#define OUT_NODE_OFF 0
#define OUT_EDGE_OFF (NN*CEMB)                    // 10240
#define OUT_MASK_OFF (OUT_EDGE_OFF + NN*NN*CEDGE) // 215040

// ---------------------------------------------------------------------------
// f16xf16->f32 GEMM on the CDNA5 WMMA path, K known at compile time.
//   D[M x N] = A[M x K] * B[K x N],  B given transposed: BT[N][K] row-major.
//   One wave32 computes a 16x32 output strip (two 16x16 N-tiles sharing the
//   A fragment -> 2 v_wmma per 6 b128 fragment loads).  The K loop fully
//   unrolls: all fragment loads are global_load_b128 at constant immediate
//   offsets from loop-invariant per-lane row pointers, and the scheduler
//   overlaps iteration k+1 loads with iteration k WMMAs (partial
//   s_wait_loadcnt) without any stage-rotation register moves.
//   A and B fragments both use contiguous K order (k = 16*half + s).  Tile cull is wave-uniform
//   so EXEC is all-ones around every WMMA; only the final f32 stores are
//   predicated on Mstore.
// ---------------------------------------------------------------------------
template <int K>
__global__ void wmma_gemm_f16f32(const h16* __restrict__ A,
                                 const h16* __restrict__ BT,
                                 float* __restrict__ D,
                                 int Mtiles, int Npairs,
                                 int Mstore, int ldd)
{
    const int wave = blockIdx.x * (blockDim.x >> 5) + (threadIdx.x >> 5);
    const int lane = threadIdx.x & 31;
    if (wave >= Mtiles * Npairs) return;            // wave-uniform
    const int mt   = wave / Npairs;
    const int np   = wave - mt * Npairs;
    const int half = lane >> 4;                     // 0: lanes 0-15, 1: 16-31
    const int l15  = lane & 15;

    // Loop-invariant per-lane row pointers with the lane-half K offset folded
    // in, so every unrolled load is base + constant immediate.
    const uint32_t* __restrict__ ap =
        (const uint32_t*)(A  + (size_t)(mt * 16 + l15) * (size_t)K) + half * 8;
    const uint32_t* __restrict__ bp0 =
        (const uint32_t*)(BT + (size_t)(np * 32 + l15) * (size_t)K) + half * 8;
    const uint32_t* __restrict__ bp1 =
        (const uint32_t*)(BT + (size_t)(np * 32 + 16 + l15) * (size_t)K) + half * 8;

    v8f acc0 = {0.f, 0.f, 0.f, 0.f, 0.f, 0.f, 0.f, 0.f};
    v8f acc1 = {0.f, 0.f, 0.f, 0.f, 0.f, 0.f, 0.f, 0.f};

#pragma unroll
    for (int k0 = 0; k0 < K; k0 += 32) {
        const int kd = k0 >> 1;                     // dwords into the row
        union { v16h v; uint32_t u[8]; } a, b0, b1;
#pragma unroll
        for (int j = 0; j < 4; ++j) {
            a.u[j]     = ap[kd + j];                // K = kA0 + {0..7}
            a.u[4 + j] = ap[kd + 4 + j];            // K = kA0 + 8 + {0..7}
        }
#pragma unroll
        for (int j = 0; j < 8; ++j) {
            b0.u[j] = bp0[kd + j];                  // K = kB0 + {0..15}
            b1.u[j] = bp1[kd + j];
        }
        acc0 = __builtin_amdgcn_wmma_f32_16x16x32_f16(
            false, a.v, false, b0.v, (short)0, acc0, false, false);
        acc1 = __builtin_amdgcn_wmma_f32_16x16x32_f16(
            false, a.v, false, b1.v, (short)0, acc1, false, false);
    }

    const int col0  = np * 32 + l15;
    const int mbase = mt * 16 + half * 8;
#pragma unroll
    for (int j = 0; j < 8; ++j) {
        const int r = mbase + j;
        if (r < Mstore) {
            D[(size_t)r * ldd + col0]      = acc0[j];
            D[(size_t)r * ldd + col0 + 16] = acc1[j];
        }
    }
}

// ---------------------------------------------------------------------------
// Weight prep: f32 conv weight [Cout][Cin][T] -> f16 BT layout [Cout][Kpad],
// column index = t*Cin + c, zero padded to Kpad.  T==1 is a plain convert.
// ---------------------------------------------------------------------------
__global__ void prep_weight(const float* __restrict__ src, h16* __restrict__ dst,
                            int Cout, int Cin, int T, int Kpad)
{
    int idx = blockIdx.x * blockDim.x + threadIdx.x;
    if (idx >= Cout * Kpad) return;
    int co  = idx / Kpad;
    int col = idx - co * Kpad;
    float v = 0.f;
    if (col < Cin * T) {
        int t = col / Cin;
        int c = col - t * Cin;
        v = src[((size_t)co * Cin + c) * T + t];
    }
    dst[idx] = (h16)v;
}

// ---------------------------------------------------------------------------
// KNN graph: pos/R/pts from affines, top-32 nearest neighbors per node
// (ascending d2, ties -> lowest index, matching lax.top_k(-d2)).
// ---------------------------------------------------------------------------
__global__ void graph_kernel(const float* __restrict__ aff,
                             const float* __restrict__ origin,
                             const float* __restrict__ voxel,
                             float* __restrict__ pos, float* __restrict__ Rm,
                             float* __restrict__ pts, int* __restrict__ eidx,
                             float* __restrict__ nbrv)
{
    int t = threadIdx.x;
    if (t < NN) {
        const float* A = aff + t * 16;
        float vs = voxel[0];
        for (int i = 0; i < 3; ++i) {
            float p = A[i * 4 + 3];
            pos[t * 3 + i] = p;
            pts[t * 3 + i] = (p - origin[i]) / vs;
            for (int j = 0; j < 3; ++j) Rm[t * 9 + i * 3 + j] = A[i * 4 + j];
        }
    }
    __syncthreads();
    if (t < NN) {
        float px = pos[t * 3], py = pos[t * 3 + 1], pz = pos[t * 3 + 2];
        float d2r[NN];
        bool  used[NN];
        for (int j = 0; j < NN; ++j) {
            float dx = px - pos[j * 3], dy = py - pos[j * 3 + 1], dz = pz - pos[j * 3 + 2];
            d2r[j]  = dx * dx + dy * dy + dz * dz + (j == t ? 1e9f : 0.f);
            used[j] = false;
        }
        for (int k = 0; k < KNBR; ++k) {
            float best = 3.4e38f; int bi = 0;
            for (int j = 0; j < NN; ++j)
                if (!used[j] && d2r[j] < best) { best = d2r[j]; bi = j; }
            used[bi] = true;
            eidx[t * KNBR + k] = bi;
            for (int i = 0; i < 3; ++i)
                nbrv[(t * KNBR + k) * 3 + i] = pos[bi * 3 + i] - pos[t * 3 + i];
        }
    }
}

__device__ __forceinline__ int clampi(int v, int lo, int hi)
{ return v < lo ? lo : (v > hi ? hi : v); }

__device__ __forceinline__ float trilerp(const float* __restrict__ g,
                                         float x, float y, float z)
{
    float xf = floorf(x), yf = floorf(y), zf = floorf(z);
    int   x0 = (int)xf,   y0 = (int)yf,   z0 = (int)zf;
    float fx = x - xf,    fy = y - yf,    fz = z - zf;
    float out = 0.f;
#pragma unroll
    for (int dx = 0; dx < 2; ++dx) {
        int ix = clampi(x0 + dx, 0, GRD - 1);
        float wx = dx ? fx : 1.f - fx;
#pragma unroll
        for (int dy = 0; dy < 2; ++dy) {
            int iy = clampi(y0 + dy, 0, GRD - 1);
            float wy = dy ? fy : 1.f - fy;
#pragma unroll
            for (int dz = 0; dz < 2; ++dz) {
                int iz = clampi(z0 + dz, 0, GRD - 1);
                float wz = dz ? fz : 1.f - fz;
                out += wx * wy * wz * g[((size_t)ix * GRD + iy) * GRD + iz];
            }
        }
    }
    return out;
}

__global__ void cube_sample(const float* __restrict__ grid,
                            const float* __restrict__ pts,
                            const float* __restrict__ Rm,
                            int node, float* __restrict__ cube)
{
    int v = blockIdx.x * blockDim.x + threadIdx.x;
    const int V = CUBE * CUBE * CUBE;
    if (v >= V) return;
    int a = v / (CUBE * CUBE);
    int r = v - a * (CUBE * CUBE);
    int b = r / CUBE, c = r - b * CUBE;
    float oa = a - (CUBE - 1) * 0.5f;
    float ob = b - (CUBE - 1) * 0.5f;
    float oc = c - (CUBE - 1) * 0.5f;
    const float* R = Rm + node * 9;
    const float* p = pts + node * 3;
    float x = p[0] + R[0] * oa + R[1] * ob + R[2] * oc;
    float y = p[1] + R[3] * oa + R[4] * ob + R[5] * oc;
    float z = p[2] + R[6] * oa + R[7] * ob + R[8] * oc;
    cube[v] = trilerp(grid, x, y, z);
}

// im2col for 3x3x3 convs: src [Din^3][Cin] -> dst [Dout^3][Kpad] f16,
// column = t*Cin + c, zero for out-of-range taps / pad columns.
template <typename T>
__global__ void im2col3(const T* __restrict__ src, h16* __restrict__ dst,
                        int Din, int Cin, int stride, int pad, int Dout, int Kpad)
{
    int idx = blockIdx.x * blockDim.x + threadIdx.x;
    int total = Dout * Dout * Dout * Kpad;
    if (idx >= total) return;
    int col = idx % Kpad;
    int vo  = idx / Kpad;
    float val = 0.f;
    if (col < 27 * Cin) {
        int t  = col / Cin;
        int cc = col - t * Cin;
        int da = t / 9, db = (t / 3) % 3, dc = t % 3;
        int oa = vo / (Dout * Dout);
        int r2 = vo - oa * Dout * Dout;
        int ob = r2 / Dout, oc = r2 - ob * Dout;
        int ia = oa * stride - pad + da;
        int ib = ob * stride - pad + db;
        int ic = oc * stride - pad + dc;
        if (ia >= 0 && ia < Din && ib >= 0 && ib < Din && ic >= 0 && ic < Din)
            val = (float)src[((size_t)(ia * Din + ib) * Din + ic) * Cin + cc];
    }
    dst[idx] = (h16)val;
}

// Per-channel instance-norm stats over M voxels (input [M][C], block = C thr).
__global__ void chan_stats(const float* __restrict__ act, int M, int C,
                           float* __restrict__ stats, float eps)
{
    int c = threadIdx.x;
    if (c >= C) return;
    float s = 0.f, q = 0.f;
    for (int v = 0; v < M; ++v) {
        float x = act[(size_t)v * C + c];
        s += x; q += x * x;
    }
    float m   = s / M;
    float var = fmaxf(q / M - m * m, 0.f);
    stats[c * 2]     = m;
    stats[c * 2 + 1] = rsqrtf(var + eps);
}

__global__ void inorm_relu_f16(const float* __restrict__ act,
                               const float* __restrict__ st,
                               const float* __restrict__ g,
                               const float* __restrict__ b,
                               h16* __restrict__ dst, int M, int C)
{
    int idx = blockIdx.x * blockDim.x + threadIdx.x;
    if (idx >= M * C) return;
    int c = idx % C;
    float y = (act[idx] - st[c * 2]) * st[c * 2 + 1] * g[c] + b[c];
    dst[idx] = (h16)fmaxf(y, 0.f);
}

// out = relu(inorm(T1) + T2) -> f16
__global__ void inorm_add_relu_f16(const float* __restrict__ T1,
                                   const float* __restrict__ T2,
                                   const float* __restrict__ st,
                                   const float* __restrict__ g,
                                   const float* __restrict__ b,
                                   h16* __restrict__ dst, int M, int C)
{
    int idx = blockIdx.x * blockDim.x + threadIdx.x;
    if (idx >= M * C) return;
    int c = idx % C;
    float y = (T1[idx] - st[c * 2]) * st[c * 2 + 1] * g[c] + b[c] + T2[idx];
    dst[idx] = (h16)fmaxf(y, 0.f);
}

// Stride-2 voxel gather (the 1x1 stride-2 shortcut conv input).
__global__ void gather_stride2(const h16* __restrict__ src, h16* __restrict__ dst,
                               int Din, int Dout, int C)
{
    int idx = blockIdx.x * blockDim.x + threadIdx.x;
    if (idx >= Dout * Dout * Dout * C) return;
    int c  = idx % C;
    int vo = idx / C;
    int oa = vo / (Dout * Dout);
    int r2 = vo - oa * Dout * Dout;
    int ob = r2 / Dout, oc = r2 - ob * Dout;
    int vin = (oa * 2 * Din + ob * 2) * Din + oc * 2;
    dst[idx] = src[(size_t)vin * C + c];
}

__global__ void ytake(const float* __restrict__ Y, h16* __restrict__ yh, int node)
{
    int c = threadIdx.x;
    if (c < CEMB) yh[node * CEMB + c] = (h16)Y[c];
}

__global__ void rect_sample(const float* __restrict__ grid,
                            const float* __restrict__ pts,
                            const float* __restrict__ nbrv,
                            float* __restrict__ rects)
{
    const int P = RECTZ * 9;
    int idx = blockIdx.x * blockDim.x + threadIdx.x;
    if (idx >= NN * KNBR * P) return;
    int e = idx / P, p = idx - e * P;
    int zi = p / 9, r = p - zi * 9, yi = r / 3, xi = r - yi * 3;
    int n = e / KNBR;
    float cx = pts[n * 3], cy = pts[n * 3 + 1], cz = pts[n * 3 + 2];
    float vx = nbrv[e * 3], vy = nbrv[e * 3 + 1], vz = nbrv[e * 3 + 2];
    float nv = sqrtf(vx * vx + vy * vy + vz * vz) + 1e-8f;
    float dx = vx / nv, dy = vy / nv, dz = vz / nv;
    float ax, ay, az;
    if (fabsf(dx) < 0.9f) { ax = 1.f; ay = 0.f; az = 0.f; }
    else                  { ax = 0.f; ay = 1.f; az = 0.f; }
    float u1x = dy * az - dz * ay, u1y = dz * ax - dx * az, u1z = dx * ay - dy * ax;
    float n1 = sqrtf(u1x * u1x + u1y * u1y + u1z * u1z) + 1e-8f;
    u1x /= n1; u1y /= n1; u1z /= n1;
    float u2x = dy * u1z - dz * u1y, u2y = dz * u1x - dx * u1z, u2z = dx * u1y - dy * u1x;
    float zv = zi - (RECTZ - 1) * 0.5f;
    float y1 = (float)(yi - 1), x1 = (float)(xi - 1);
    float X = cx + zv * dx + y1 * u1x + x1 * u2x;
    float Y = cy + zv * dy + y1 * u1y + x1 * u2y;
    float Z = cz + zv * dz + y1 * u1z + x1 * u2z;
    rects[idx] = trilerp(grid, X, Y, Z);
}

// 1x3x3 conv, 1->128 ch: h[e][c*17+z] = sum_{yi,xi} w[c][yi][xi]*rect[e][z][yi][xi]
__global__ void rect_conv(const float* __restrict__ rects,
                          const float* __restrict__ w,
                          float* __restrict__ h)
{
    int idx = blockIdx.x * blockDim.x + threadIdx.x;
    if (idx >= NN * KNBR * CEDGE * RECTZ) return;
    int e = idx / (CEDGE * RECTZ);
    int r = idx - e * (CEDGE * RECTZ);
    int c = r / RECTZ, z = r - c * RECTZ;
    float s = 0.f;
#pragma unroll
    for (int q = 0; q < 9; ++q)
        s += w[c * 9 + q] * rects[e * (RECTZ * 9) + z * 9 + q];
    h[(size_t)e * LNDIM + c * RECTZ + z] = s;
}

__global__ void ln_stats(const float* __restrict__ h, float* __restrict__ st)
{
    __shared__ float ss[256], sq[256];
    int e = blockIdx.x, t = threadIdx.x;
    float s = 0.f, q = 0.f;
    for (int i = t; i < LNDIM; i += 256) {
        float x = h[(size_t)e * LNDIM + i];
        s += x; q += x * x;
    }
    ss[t] = s; sq[t] = q; __syncthreads();
    for (int w = 128; w > 0; w >>= 1) {
        if (t < w) { ss[t] += ss[t + w]; sq[t] += sq[t + w]; }
        __syncthreads();
    }
    if (t == 0) {
        float m   = ss[0] / LNDIM;
        float var = fmaxf(sq[0] / LNDIM - m * m, 0.f);
        st[e * 2]     = m;
        st[e * 2 + 1] = rsqrtf(var + 1e-5f);
    }
}

__global__ void ln_relu_f16(const float* __restrict__ h, const float* __restrict__ st,
                            const float* __restrict__ g, const float* __restrict__ b,
                            h16* __restrict__ dst)
{
    int idx = blockIdx.x * blockDim.x + threadIdx.x;
    if (idx >= NN * KNBR * LNDIM) return;
    int e = idx / LNDIM, c = idx - e * LNDIM;
    float y = (h[idx] - st[e * 2]) * st[e * 2 + 1] * g[c] + b[c];
    dst[idx] = (h16)fmaxf(y, 0.f);
}

__global__ void out_init(float* __restrict__ out)
{
    int idx = blockIdx.x * blockDim.x + threadIdx.x;
    if (idx < NN * NN * CEDGE) out[OUT_EDGE_OFF + idx] = 0.f;
    if (idx < NN)              out[OUT_MASK_OFF + idx] = 1.f;
}

__global__ void scatter_edges(const float* __restrict__ eout,
                              const int* __restrict__ eidx,
                              float* __restrict__ out)
{
    int idx = blockIdx.x * blockDim.x + threadIdx.x;
    if (idx >= NN * KNBR * CEDGE) return;
    int e = idx >> 7, c = idx & 127;
    int n = e / KNBR;
    int col = eidx[e];
    out[OUT_EDGE_OFF + ((size_t)(n * NN + col)) * CEDGE + c] = eout[idx];
}

// ---------------------------------------------------------------------------
extern "C" void kernel_launch(void* const* d_in, const int* in_sizes, int n_in,
                              void* d_out, int out_size, void* d_ws, size_t ws_size,
                              hipStream_t stream)
{
    (void)in_sizes; (void)n_in; (void)out_size; (void)ws_size;

    const float* aff    = (const float*)d_in[0];
    const float* grid   = (const float*)d_in[1];
    const float* origin = (const float*)d_in[2];
    const float* voxel  = (const float*)d_in[3];
    const float* cb_w   = (const float*)d_in[4];
    const float* cb_g   = (const float*)d_in[5];
    const float* cb_b   = (const float*)d_in[6];
    const float *w1[4], *g1[4], *b1[4], *w2[4], *g2[4], *b2[4],
                *w3[4], *g3[4], *b3[4], *wsk[4];
    for (int i = 0; i < 4; ++i) {
        int o = 7 + i * 10;
        w1[i]  = (const float*)d_in[o + 0];
        g1[i]  = (const float*)d_in[o + 1];
        b1[i]  = (const float*)d_in[o + 2];
        w2[i]  = (const float*)d_in[o + 3];
        g2[i]  = (const float*)d_in[o + 4];
        b2[i]  = (const float*)d_in[o + 5];
        w3[i]  = (const float*)d_in[o + 6];
        g3[i]  = (const float*)d_in[o + 7];
        b3[i]  = (const float*)d_in[o + 8];
        wsk[i] = (const float*)d_in[o + 9];
    }
    const float* final_w = (const float*)d_in[47];
    const float* node_w  = (const float*)d_in[48];
    const float* rect_w  = (const float*)d_in[49];
    const float* ln_g    = (const float*)d_in[50];
    const float* ln_b    = (const float*)d_in[51];
    const float* edge_w  = (const float*)d_in[52];
    float* out = (float*)d_out;

    // --- workspace bump allocator ---
    char*  base = (char*)d_ws;
    size_t off  = 0;
    auto alloc = [&](size_t bytes) -> void* {
        void* p = base + off;
        off += (bytes + 255) & ~(size_t)255;
        return p;
    };

    h16* cbT = (h16*)alloc(256 * 32 * 2);
    h16 *w1h[4], *w2h[4], *w3h[4], *wsh[4];
    for (int i = 0; i < 4; ++i) {
        w1h[i] = (h16*)alloc(64 * 256 * 2);
        w2h[i] = (h16*)alloc(64 * 1728 * 2);
        w3h[i] = (h16*)alloc(256 * 64 * 2);
        wsh[i] = (h16*)alloc(256 * 256 * 2);
    }
    h16* finalh = (h16*)alloc(256 * 2048 * 2);
    h16* nodeh  = (h16*)alloc(256 * 256 * 2);
    h16* edgeh  = (h16*)alloc(128 * 2176 * 2);

    float* pos  = (float*)alloc(NN * 3 * 4);
    float* Rm   = (float*)alloc(NN * 9 * 4);
    float* ptsb = (float*)alloc(NN * 3 * 4);
    int*   eidx = (int*)alloc(NN * KNBR * 4);
    float* nbrv = (float*)alloc(NN * KNBR * 3 * 4);
    h16*   yallh = (h16*)alloc(48 * 256 * 2);
    float* stats = (float*)alloc(256 * 2 * 4);

    float* cube = (float*)alloc((size_t)12176 * 4);
    h16*   Xh   = (h16*)alloc((size_t)12176 * 256 * 2);
    float* Yb   = (float*)alloc((size_t)12176 * 256 * 4);
    h16*   Ph   = (h16*)alloc((size_t)12176 * 64 * 2);
    h16*   Xcol = (h16*)alloc((size_t)1728 * 1728 * 2);
    float* T1   = (float*)alloc((size_t)1728 * 256 * 4);
    float* T2   = (float*)alloc((size_t)1728 * 256 * 4);

    float* rects = (float*)alloc((size_t)NN * KNBR * RECTZ * 9 * 4);
    float* hf    = (float*)alloc((size_t)NN * KNBR * LNDIM * 4);
    h16*   hh    = (h16*)alloc((size_t)NN * KNBR * LNDIM * 2);
    float* lnst  = (float*)alloc((size_t)NN * KNBR * 2 * 4);
    float* eoutb = (float*)alloc((size_t)NN * KNBR * CEDGE * 4);

    auto prep = [&](const float* s, h16* d, int Co, int Ci, int T, int Kp) {
        int tot = Co * Kp;
        prep_weight<<<CEILDIV(tot, 256), 256, 0, stream>>>(s, d, Co, Ci, T, Kp);
    };
    // N is always a multiple of 32 in this pipeline -> Npairs = Ntiles/2.
    // K is dispatched to a compile-time instantiation (full unroll).
    auto gemm = [&](const h16* A, const h16* BT, float* D,
                    int Mtiles, int Ntiles, int K, int Mstore, int ldd) {
        int waves  = Mtiles * (Ntiles / 2);
        int blocks = CEILDIV(waves, 8);
        int np     = Ntiles / 2;
        switch (K) {
        case 32:
            wmma_gemm_f16f32<32><<<blocks, 256, 0, stream>>>(A, BT, D, Mtiles, np, Mstore, ldd);
            break;
        case 64:
            wmma_gemm_f16f32<64><<<blocks, 256, 0, stream>>>(A, BT, D, Mtiles, np, Mstore, ldd);
            break;
        case 256:
            wmma_gemm_f16f32<256><<<blocks, 256, 0, stream>>>(A, BT, D, Mtiles, np, Mstore, ldd);
            break;
        case 1728:
            wmma_gemm_f16f32<1728><<<blocks, 256, 0, stream>>>(A, BT, D, Mtiles, np, Mstore, ldd);
            break;
        case 2048:
            wmma_gemm_f16f32<2048><<<blocks, 256, 0, stream>>>(A, BT, D, Mtiles, np, Mstore, ldd);
            break;
        case 2176:
            wmma_gemm_f16f32<2176><<<blocks, 256, 0, stream>>>(A, BT, D, Mtiles, np, Mstore, ldd);
            break;
        default:
            break; // unreachable in this pipeline
        }
    };

    // --- weights -> f16 BT layout ---
    prep(cb_w, cbT, 256, 1, 27, 32);
    for (int i = 0; i < 4; ++i) {
        prep(w1[i],  w1h[i], 64, 256, 1, 256);
        prep(w2[i],  w2h[i], 64, 64, 27, 1728);
        prep(w3[i],  w3h[i], 256, 64, 1, 64);
        prep(wsk[i], wsh[i], 256, 256, 1, 256);
    }
    prep(final_w, finalh, 256, 256, 8, 2048);
    prep(node_w,  nodeh,  256, 256, 1, 256);
    prep(edge_w,  edgeh,  128, 2176, 1, 2176);

    // --- KNN graph ---
    graph_kernel<<<1, 64, 0, stream>>>(aff, origin, voxel, pos, Rm, ptsb, eidx, nbrv);

    // --- per-node cube CNN ---
    const int Din[4]   = {23, 12, 6, 3};
    const int Dout_[4] = {12, 6, 3, 2};
    const int Min[4]   = {12167, 1728, 216, 27};
    const int Minp[4]  = {12176, 1728, 224, 32};
    const int Mout[4]  = {1728, 216, 27, 8};
    const int Moutp[4] = {1728, 224, 32, 16};

    for (int node = 0; node < NN; ++node) {
        cube_sample<<<CEILDIV(12167, 256), 256, 0, stream>>>(grid, ptsb, Rm, node, cube);

        // conv1: im2col (K=27 -> 32) + WMMA GEMM, then inorm+relu -> f16 x
        im2col3<float><<<CEILDIV(12167 * 32, 256), 256, 0, stream>>>(
            cube, Xcol, 23, 1, 1, 1, 23, 32);
        gemm(Xcol, cbT, Yb, 761, 16, 32, 12167, 256);
        chan_stats<<<1, 256, 0, stream>>>(Yb, 12167, 256, stats, 1e-5f);
        inorm_relu_f16<<<CEILDIV(12167 * 256, 256), 256, 0, stream>>>(
            Yb, stats, cb_g, cb_b, Xh, 12167, 256);

        for (int bl = 0; bl < 4; ++bl) {
            int di = Din[bl], doo = Dout_[bl];
            int mi = Min[bl], mip = Minp[bl];
            int mo = Mout[bl], mop = Moutp[bl];

            // w1: 1x1 conv 256->64 as GEMM
            gemm(Xh, w1h[bl], Yb, mip / 16, 4, 256, mi, 64);
            chan_stats<<<1, 64, 0, stream>>>(Yb, mi, 64, stats, 1e-5f);
            inorm_relu_f16<<<CEILDIV(mi * 64, 256), 256, 0, stream>>>(
                Yb, stats, g1[bl], b1[bl], Ph, mi, 64);

            // w2: 3x3x3 s2 p1 conv via im2col (K = 27*64 = 1728) + GEMM
            im2col3<h16><<<CEILDIV(mo * 1728, 256), 256, 0, stream>>>(
                Ph, Xcol, di, 64, 2, 1, doo, 1728);
            gemm(Xcol, w2h[bl], Yb, mop / 16, 4, 1728, mo, 64);
            chan_stats<<<1, 64, 0, stream>>>(Yb, mo, 64, stats, 1e-5f);
            inorm_relu_f16<<<CEILDIV(mo * 64, 256), 256, 0, stream>>>(
                Yb, stats, g2[bl], b2[bl], Ph, mo, 64);

            // w3: 1x1 conv 64->256
            gemm(Ph, w3h[bl], T1, mop / 16, 16, 64, mo, 256);

            // shortcut: stride-2 gather of x, then 1x1 conv 256->256
            gather_stride2<<<CEILDIV(mo * 256, 256), 256, 0, stream>>>(
                Xh, Xcol, di, doo, 256);
            gemm(Xcol, wsh[bl], T2, mop / 16, 16, 256, mo, 256);

            // relu(inorm(w3) + shortcut) -> new f16 x
            chan_stats<<<1, 256, 0, stream>>>(T1, mo, 256, stats, 1e-5f);
            inorm_add_relu_f16<<<CEILDIV(mo * 256, 256), 256, 0, stream>>>(
                T1, T2, stats, g3[bl], b3[bl], Xh, mo, 256);
        }

        // final 2x2x2 conv == M=1 GEMM over K=2048 (x already flat t-major/c-minor)
        gemm(Xh, finalh, Yb, 1, 16, 2048, 1, 256);
        ytake<<<1, 256, 0, stream>>>(Yb, yallh, node);
    }

    // node_repr = y @ node_w.T  (directly into d_out rows 0..39)
    gemm(yallh, nodeh, out + OUT_NODE_OFF, 3, 16, 256, NN, 256);

    // --- edge branch ---
    rect_sample<<<CEILDIV(NN * KNBR * RECTZ * 9, 256), 256, 0, stream>>>(
        grid, ptsb, nbrv, rects);
    rect_conv<<<CEILDIV(NN * KNBR * CEDGE * RECTZ, 256), 256, 0, stream>>>(
        rects, rect_w, hf);
    ln_stats<<<NN * KNBR, 256, 0, stream>>>(hf, lnst);
    ln_relu_f16<<<CEILDIV(NN * KNBR * LNDIM, 256), 256, 0, stream>>>(
        hf, lnst, ln_g, ln_b, hh);
    gemm(hh, edgeh, eoutb, 80, 8, 2176, NN * KNBR, 128);

    out_init<<<CEILDIV(NN * NN * CEDGE, 256), 256, 0, stream>>>(out);
    scatter_edges<<<CEILDIV(NN * KNBR * CEDGE, 256), 256, 0, stream>>>(eoutb, eidx, out);
}
